// GNN_38766374814174
// MI455X (gfx1250) — matrix operand order repaired and checked by reading, not compiled, not run on hardware
//
#include <hip/hip_runtime.h>
#include <hip/hip_bf16.h>

typedef __attribute__((ext_vector_type(16))) __bf16 v16bf;
typedef __attribute__((ext_vector_type(8)))  __bf16 v8bf;
typedef __attribute__((ext_vector_type(8)))  float  v8f;

#define N_NODES 100000
#define M_EDGES 800000
#define DFEAT   128
#define HID     256

// Branch-free exact-GELU: Abramowitz-Stegun 7.1.26 erf (|err| ~ 1.5e-7),
// single v_rcp_f32 + v_exp_f32, no exec-mask divergence in the hot loop.
__device__ __forceinline__ float gelu_exact(float x) {
    const float z  = x * 0.70710678118654752f;
    const float az = __builtin_fabsf(z);
    const float t  = __builtin_amdgcn_rcpf(1.0f + 0.3275911f * az);
    float p = 1.061405429f;
    p = p * t - 1.453152027f;
    p = p * t + 1.421413741f;
    p = p * t - 0.284496736f;
    p = p * t + 0.254829592f;
    p = p * t;
    const float e    = __expf(-az * az);
    const float erfa = 1.0f - p * e;
    const float erfz = __builtin_copysignf(erfa, z);
    return 0.5f * x * (1.0f + erfz);
}

__device__ __forceinline__ v8f wmma_bf16(v16bf a, v16bf b, v8f c) {
    return __builtin_amdgcn_wmma_f32_16x16x32_bf16(false, a, false, b,
                                                   (short)0, c, false, false);
}

// Flat pointers into LDS carry the LDS byte offset in their low 32 bits
// (addrspacecast as3->generic builds {SHARED_BASE_HI, lds_offset}).
__device__ __forceinline__ uint32_t lds_off32(const void* p) {
    return (uint32_t)(uintptr_t)p;
}

// GLOBAL_LOAD_ASYNC_TO_LDS_B128 (GVS mode): per-lane 16B memory->LDS copy,
// tracked by ASYNCcnt. vdst = LDS byte offset, vaddr = 32-bit byte offset,
// saddr = 64-bit base.
__device__ __forceinline__ void async_cp16(uint32_t lds_byte_off,
                                           const void* sbase, uint32_t voff) {
    asm volatile("global_load_async_to_lds_b128 %0, %1, %2"
                 :: "v"(lds_byte_off), "v"(voff), "s"(sbase)
                 : "memory");
}

template<int N>
__device__ __forceinline__ void wait_async() {
    asm volatile("s_wait_asynccnt %0" :: "n"(N) : "memory");
}

// Stage SWEEPS*4096 bytes (contiguous) from gsrc into LDS, 256 threads,
// 16B per thread per sweep. Each wave issues SWEEPS async instructions.
template<int SWEEPS>
__device__ __forceinline__ void stage_weights(const __bf16* gsrc,
                                              const __bf16* ldst, int tid) {
    const uint32_t base = lds_off32(ldst);
#pragma unroll
    for (int c = 0; c < SWEEPS; ++c) {
        const uint32_t off = (uint32_t)(tid * 16 + c * 4096);
        async_cp16(base + off, gsrc, off);
    }
}

// out[n*K + k] = (bf16) W[k*Ncols + n]   (transpose + convert)
__global__ void convert_transpose_kernel(const float* __restrict__ W,
                                         __bf16* __restrict__ out,
                                         int K, int Ncols) {
    int idx = blockIdx.x * blockDim.x + threadIdx.x;
    if (idx >= K * Ncols) return;
    int nn = idx / K;
    int kk = idx - nn * K;
    out[idx] = (__bf16)W[(size_t)kk * Ncols + nn];
}

// Build one layer-2 A fragment (16x32 bf16) from a 128-wide hidden slice.
__device__ __forceinline__ v16bf a_frag_from_hid(const __bf16* hp, int kb_local, int g) {
    const int K0 = kb_local * 32 + g * 8;
    v8bf lo = *(const v8bf*)(hp + K0);
    v8bf hi = *(const v8bf*)(hp + K0 + 16);
    v16bf f;
#pragma unroll
    for (int h = 0; h < 8; ++h) { f[h] = lo[h]; f[h + 8] = hi[h]; }
    return f;
}

// ---------------------------------------------------------------------------
// Edge kernel: 8 waves/block, 16 edges per wave (128 edges/block).
// x = [V[sender] | V[receiver] | E[e]] (384) -> 256 (GELU) -> 128
// Weight tiles double-buffered in LDS via async copies; B-frags via ds loads.
// Hidden is produced in two K-halves so 8 waves fit in < 64 KB of LDS.
// ---------------------------------------------------------------------------
__global__ __launch_bounds__(256)
void edge_mlp_kernel(const float* __restrict__ V,
                     const float* __restrict__ E,
                     const int*   __restrict__ edges,   // [M][2]
                     const __bf16* __restrict__ W1T,    // [256][384]
                     const float* __restrict__ b1,      // [256]
                     const __bf16* __restrict__ W2T,    // [128][256]
                     const float* __restrict__ b2,      // [128]
                     float* __restrict__ edge_out,      // [M][128]
                     float* __restrict__ seg,           // [N][128] accumulator
                     float* __restrict__ cnt)           // [N]
{
    __shared__ __align__(32) __bf16 hid[8][16][128];     // 32 KB (one K-half)
    __shared__ __align__(32) __bf16 wstage[2][6144];     // 2 x 12 KB

    const int tid  = threadIdx.x;
    const int wave = tid >> 5;
    const int lane = tid & 31;
    const int m    = lane & 15;   // edge row in tile / weight column in tile
    const int g    = lane >> 4;
    const int e0   = blockIdx.x * 128 + wave * 16;

    // kick off the first layer-1 weight tile DMA before doing the gathers
    stage_weights<3>(W1T, wstage[0], tid);

    // in-degree counts: one lane per edge
    if (g == 0) {
        atomicAdd(&cnt[edges[2 * (e0 + m) + 1]], 1.0f);
    }

    // ---- build A fragments: 16x384 bf16 as 12 fragments of 16x32 ----
    const int e = e0 + m;
    const int s = edges[2 * e + 0];
    const int r = edges[2 * e + 1];
    const float* src0 = V + (size_t)s * DFEAT;
    const float* src1 = V + (size_t)r * DFEAT;
    const float* src2 = E + (size_t)e * DFEAT;

    v16bf a[12];
#pragma unroll
    for (int kb = 0; kb < 12; ++kb) {
        const int K0 = kb * 32 + g * 8;        // halves 0..7  -> K0..K0+7
        const int K1 = K0 + 16;                // halves 8..15 -> K1..K1+7
        const float* p0 = (K0 < 128) ? (src0 + K0)
                        : (K0 < 256) ? (src1 + (K0 - 128))
                                     : (src2 + (K0 - 256));
        const float* p1 = (K1 < 128) ? (src0 + K1)
                        : (K1 < 256) ? (src1 + (K1 - 128))
                                     : (src2 + (K1 - 256));
        float4 u0 = ((const float4*)p0)[0];
        float4 u1 = ((const float4*)p0)[1];
        float4 u2 = ((const float4*)p1)[0];
        float4 u3 = ((const float4*)p1)[1];
        a[kb][0]  = (__bf16)u0.x; a[kb][1]  = (__bf16)u0.y;
        a[kb][2]  = (__bf16)u0.z; a[kb][3]  = (__bf16)u0.w;
        a[kb][4]  = (__bf16)u1.x; a[kb][5]  = (__bf16)u1.y;
        a[kb][6]  = (__bf16)u1.z; a[kb][7]  = (__bf16)u1.w;
        a[kb][8]  = (__bf16)u2.x; a[kb][9]  = (__bf16)u2.y;
        a[kb][10] = (__bf16)u2.z; a[kb][11] = (__bf16)u2.w;
        a[kb][12] = (__bf16)u3.x; a[kb][13] = (__bf16)u3.y;
        a[kb][14] = (__bf16)u3.z; a[kb][15] = (__bf16)u3.w;
    }

    // ---- layer 1: 16 output tiles of 16x16, K = 384; hidden in 2 halves ----
    v16bf a2[8];
    const __bf16* hp = &hid[wave][m][0];

    for (int t = 0; t < 16; ++t) {
        if (t < 15) {
            stage_weights<3>(W1T + (size_t)(t + 1) * 6144, wstage[(t + 1) & 1], tid);
            wait_async<3>();   // previous stage complete, new one in flight
        } else {
            wait_async<0>();
        }
        __syncthreads();       // all waves' DMA portions visible

        const float bias = b1[t * 16 + m];
        v8f c;
#pragma unroll
        for (int i = 0; i < 8; ++i) c[i] = bias;
        const __bf16* wp = &wstage[t & 1][m * 384 + g * 8];
#pragma unroll
        for (int kb = 0; kb < 12; ++kb) {
            v8bf blo = *(const v8bf*)(wp + kb * 32);
            v8bf bhi = *(const v8bf*)(wp + kb * 32 + 16);
            v16bf b = __builtin_shufflevector(blo, bhi, 0,1,2,3,4,5,6,7,8,9,10,11,12,13,14,15);
            c = wmma_bf16(a[kb], b, c);
        }
#pragma unroll
        for (int rr = 0; rr < 8; ++rr) {
            hid[wave][rr + 8 * g][(t & 7) * 16 + m] = (__bf16)gelu_exact(c[rr]);
        }
        __syncthreads();       // done reading this buffer before it is refilled

        // lift finished hidden K-half into layer-2 A fragments (wave-local)
        if (t == 7) {
#pragma unroll
            for (int kb = 0; kb < 4; ++kb) a2[kb] = a_frag_from_hid(hp, kb, g);
        }
    }
#pragma unroll
    for (int kb = 0; kb < 4; ++kb) a2[kb + 4] = a_frag_from_hid(hp, kb, g);

    // ---- stage first layer-2 tile ----
    stage_weights<2>(W2T, wstage[0], tid);

    int rrs[8];
#pragma unroll
    for (int rr = 0; rr < 8; ++rr) rrs[rr] = edges[2 * (e0 + rr + 8 * g) + 1];

    // ---- layer 2: 8 output tiles of 16x16, K = 256 ----
    for (int t2 = 0; t2 < 8; ++t2) {
        if (t2 < 7) {
            stage_weights<2>(W2T + (size_t)(t2 + 1) * 4096, wstage[(t2 + 1) & 1], tid);
            wait_async<2>();
        } else {
            wait_async<0>();
        }
        __syncthreads();

        const int col = t2 * 16 + m;
        const float bias = b2[col];
        v8f c;
#pragma unroll
        for (int i = 0; i < 8; ++i) c[i] = bias;
        const __bf16* wp = &wstage[t2 & 1][m * 256 + g * 8];
#pragma unroll
        for (int kb = 0; kb < 8; ++kb) {
            v8bf blo = *(const v8bf*)(wp + kb * 32);
            v8bf bhi = *(const v8bf*)(wp + kb * 32 + 16);
            v16bf b = __builtin_shufflevector(blo, bhi, 0,1,2,3,4,5,6,7,8,9,10,11,12,13,14,15);
            c = wmma_bf16(a2[kb], b, c);
        }
#pragma unroll
        for (int rr = 0; rr < 8; ++rr) {
            const int row = rr + 8 * g;
            const int ee  = e0 + row;
            edge_out[(size_t)ee * 128 + col] = c[rr];
            atomicAdd(&seg[(size_t)rrs[rr] * 128 + col], c[rr]);
        }
        __syncthreads();
    }
}

// ---------------------------------------------------------------------------
// Node kernel: x = [V[n] | seg[n]/max(cnt,1)] (256) -> 256 (GELU) -> 128
// seg lives in the node region of d_out and is overwritten with node_emb.
// ---------------------------------------------------------------------------
__global__ __launch_bounds__(256)
void node_mlp_kernel(const float* __restrict__ V,
                     const __bf16* __restrict__ Wn1T,  // [256][256]
                     const float* __restrict__ bn1,
                     const __bf16* __restrict__ Wn2T,  // [128][256]
                     const float* __restrict__ bn2,
                     const float* __restrict__ cnt,
                     float* __restrict__ node_out)     // [N][128]
{
    __shared__ __align__(32) __bf16 hid[8][16][128];     // 32 KB (one K-half)
    __shared__ __align__(32) __bf16 wstage[2][4096];     // 2 x 8 KB

    const int tid  = threadIdx.x;
    const int wave = tid >> 5;
    const int lane = tid & 31;
    const int m    = lane & 15;
    const int g    = lane >> 4;
    const int n0   = blockIdx.x * 128 + wave * 16;

    stage_weights<2>(Wn1T, wstage[0], tid);

    int nr = n0 + m;
    if (nr >= N_NODES) nr = N_NODES - 1;   // clamp loads, mask stores later

    const float inv = 1.0f / fmaxf(cnt[nr], 1.0f);
    const float* pv = V        + (size_t)nr * DFEAT;
    const float* ps = node_out + (size_t)nr * DFEAT;   // seg_sum accumulator

    v16bf a[8];
#pragma unroll
    for (int kb = 0; kb < 8; ++kb) {
        const int K0 = kb * 32 + g * 8;   // each 8-group lands on one source
        const int K1 = K0 + 16;
        const bool lo_v = (K0 < 128), hi_v = (K1 < 128);
        const float* p0 = lo_v ? (pv + K0) : (ps + (K0 - 128));
        const float* p1 = hi_v ? (pv + K1) : (ps + (K1 - 128));
        const float s0 = lo_v ? 1.0f : inv;
        const float s1 = hi_v ? 1.0f : inv;
        float4 u0 = ((const float4*)p0)[0];
        float4 u1 = ((const float4*)p0)[1];
        float4 u2 = ((const float4*)p1)[0];
        float4 u3 = ((const float4*)p1)[1];
        a[kb][0]  = (__bf16)(u0.x * s0); a[kb][1]  = (__bf16)(u0.y * s0);
        a[kb][2]  = (__bf16)(u0.z * s0); a[kb][3]  = (__bf16)(u0.w * s0);
        a[kb][4]  = (__bf16)(u1.x * s0); a[kb][5]  = (__bf16)(u1.y * s0);
        a[kb][6]  = (__bf16)(u1.z * s0); a[kb][7]  = (__bf16)(u1.w * s0);
        a[kb][8]  = (__bf16)(u2.x * s1); a[kb][9]  = (__bf16)(u2.y * s1);
        a[kb][10] = (__bf16)(u2.z * s1); a[kb][11] = (__bf16)(u2.w * s1);
        a[kb][12] = (__bf16)(u3.x * s1); a[kb][13] = (__bf16)(u3.y * s1);
        a[kb][14] = (__bf16)(u3.z * s1); a[kb][15] = (__bf16)(u3.w * s1);
    }

    // ---- layer 1: 16 tiles, K = 256; hidden in 2 halves ----
    v16bf a2[8];
    const __bf16* hp = &hid[wave][m][0];

    for (int t = 0; t < 16; ++t) {
        if (t < 15) {
            stage_weights<2>(Wn1T + (size_t)(t + 1) * 4096, wstage[(t + 1) & 1], tid);
            wait_async<2>();
        } else {
            wait_async<0>();
        }
        __syncthreads();

        const float bias = bn1[t * 16 + m];
        v8f c;
#pragma unroll
        for (int i = 0; i < 8; ++i) c[i] = bias;
        const __bf16* wp = &wstage[t & 1][m * 256 + g * 8];
#pragma unroll
        for (int kb = 0; kb < 8; ++kb) {
            v8bf blo = *(const v8bf*)(wp + kb * 32);
            v8bf bhi = *(const v8bf*)(wp + kb * 32 + 16);
            v16bf b = __builtin_shufflevector(blo, bhi, 0,1,2,3,4,5,6,7,8,9,10,11,12,13,14,15);
            c = wmma_bf16(a[kb], b, c);
        }
#pragma unroll
        for (int rr = 0; rr < 8; ++rr) {
            hid[wave][rr + 8 * g][(t & 7) * 16 + m] = (__bf16)gelu_exact(c[rr]);
        }
        __syncthreads();

        if (t == 7) {
#pragma unroll
            for (int kb = 0; kb < 4; ++kb) a2[kb] = a_frag_from_hid(hp, kb, g);
        }
    }
#pragma unroll
    for (int kb = 0; kb < 4; ++kb) a2[kb + 4] = a_frag_from_hid(hp, kb, g);

    stage_weights<2>(Wn2T, wstage[0], tid);

    // ---- layer 2: 8 tiles, K = 256; overwrite accumulator with node_emb ----
    for (int t2 = 0; t2 < 8; ++t2) {
        if (t2 < 7) {
            stage_weights<2>(Wn2T + (size_t)(t2 + 1) * 4096, wstage[(t2 + 1) & 1], tid);
            wait_async<2>();
        } else {
            wait_async<0>();
        }
        __syncthreads();

        const int col = t2 * 16 + m;
        const float bias = bn2[col];
        v8f c;
#pragma unroll
        for (int i = 0; i < 8; ++i) c[i] = bias;
        const __bf16* wp = &wstage[t2 & 1][m * 256 + g * 8];
#pragma unroll
        for (int kb = 0; kb < 8; ++kb) {
            v8bf blo = *(const v8bf*)(wp + kb * 32);
            v8bf bhi = *(const v8bf*)(wp + kb * 32 + 16);
            v16bf b = __builtin_shufflevector(blo, bhi, 0,1,2,3,4,5,6,7,8,9,10,11,12,13,14,15);
            c = wmma_bf16(a2[kb], b, c);
        }
#pragma unroll
        for (int rr = 0; rr < 8; ++rr) {
            const int node = n0 + rr + 8 * g;
            if (node < N_NODES) node_out[(size_t)node * 128 + col] = c[rr];
        }
        __syncthreads();
    }
}

extern "C" void kernel_launch(void* const* d_in, const int* in_sizes, int n_in,
                              void* d_out, int out_size, void* d_ws, size_t ws_size,
                              hipStream_t stream) {
    const float* V   = (const float*)d_in[0];
    const float* E   = (const float*)d_in[1];
    const int*   edg = (const int*)  d_in[2];
    const float* We1 = (const float*)d_in[3];
    const float* be1 = (const float*)d_in[4];
    const float* We2 = (const float*)d_in[5];
    const float* be2 = (const float*)d_in[6];
    const float* Wn1 = (const float*)d_in[7];
    const float* bn1 = (const float*)d_in[8];
    const float* Wn2 = (const float*)d_in[9];
    const float* bn2 = (const float*)d_in[10];

    float* node_out = (float*)d_out;                           // [N][128]
    float* edge_out = (float*)d_out + (size_t)N_NODES * 128;   // [M][128]

    char* ws = (char*)d_ws;
    __bf16* W1T  = (__bf16*)ws; ws += (size_t)256 * 384 * sizeof(__bf16);
    __bf16* W2T  = (__bf16*)ws; ws += (size_t)128 * 256 * sizeof(__bf16);
    __bf16* Wn1T = (__bf16*)ws; ws += (size_t)256 * 256 * sizeof(__bf16);
    __bf16* Wn2T = (__bf16*)ws; ws += (size_t)128 * 256 * sizeof(__bf16);
    float*  cnt  = (float*)ws;  ws += (size_t)N_NODES * sizeof(float);

    // zero the scatter accumulators (node region of d_out doubles as seg_sum)
    hipMemsetAsync(node_out, 0, (size_t)N_NODES * 128 * sizeof(float), stream);
    hipMemsetAsync(cnt,      0, (size_t)N_NODES *       sizeof(float), stream);

    // convert + transpose weights to bf16
    convert_transpose_kernel<<<(384 * 256 + 255) / 256, 256, 0, stream>>>(We1, W1T, 384, 256);
    convert_transpose_kernel<<<(256 * 128 + 255) / 256, 256, 0, stream>>>(We2, W2T, 256, 128);
    convert_transpose_kernel<<<(256 * 256 + 255) / 256, 256, 0, stream>>>(Wn1, Wn1T, 256, 256);
    convert_transpose_kernel<<<(256 * 128 + 255) / 256, 256, 0, stream>>>(Wn2, Wn2T, 256, 128);

    // edge MLP + scatter: M/128 = 6250 blocks exactly
    edge_mlp_kernel<<<M_EDGES / 128, 256, 0, stream>>>(
        V, E, edg, W1T, be1, W2T, be2, edge_out, node_out, cnt);

    // node MLP: ceil(N/128) blocks
    node_mlp_kernel<<<(N_NODES + 127) / 128, 256, 0, stream>>>(
        V, Wn1T, bn1, Wn2T, bn2, cnt, node_out);
}
